// MaskedAutoencoderViTRoPE_6244882448865
// MI455X (gfx1250) — compile-verified, matrix operand order checked by reading, not run on hardware
//
#include <hip/hip_runtime.h>
#include <hip/hip_bf16.h>
#include <math.h>

// ---------------------------------------------------------------------------
// Types for CDNA5 WMMA (gfx1250, wave32)
// ---------------------------------------------------------------------------
typedef __attribute__((ext_vector_type(16))) __bf16 v16bf;
typedef __attribute__((ext_vector_type(8)))  float  v8f;

static __device__ inline v8f zero8() {
  v8f z = {0.f, 0.f, 0.f, 0.f, 0.f, 0.f, 0.f, 0.f};
  return z;
}

static __device__ inline v8f wmma_bf16(v16bf a, v16bf b, v8f c) {
  // D(16x16,f32) = A(16x32,bf16) x B(32x16,bf16) + C
  return __builtin_amdgcn_wmma_f32_16x16x32_bf16(
      /*neg_a=*/false, a, /*neg_b=*/false, b,
      /*c_mod=*/(short)0, c, /*reuse_a=*/false, /*reuse_b=*/false);
}

// Row-major fragment load (used for A, and for B==W^T where W is N x K row
// major: lane -> matrix row/col index, vector elems -> K).
// 16-bit A-matrix 16x32 layout (ISA 7.12.2):
//   lanes 0-15 : row = lane,    elems 0..7 -> K 0..7,  elems 8..15 -> K 16..23
//   lanes16-31 : row = lane-16, elems 0..7 -> K 8..15, elems 8..15 -> K 24..31
// stride must be a multiple of 8 bf16 (16 B) and base 16-B aligned.
static __device__ inline v16bf load_frag_rm(const __bf16* base, int stride) {
  int lane = threadIdx.x & 31;
  int row  = lane & 15;
  int kh   = (lane >> 4) << 3;  // 0 or 8
  const __bf16* p = base + row * stride + kh;
  v16bf f;
  *reinterpret_cast<uint4*>(&f)       = *reinterpret_cast<const uint4*>(p);
  *(reinterpret_cast<uint4*>(&f) + 1) = *reinterpret_cast<const uint4*>(p + 16);
  return f;
}

// Column-major fragment load for B from a row-major K x N tile in LDS:
// lane -> output column, vector elems -> K rows (same K striping as above).
static __device__ inline v16bf load_frag_cm(const __bf16* base, int stride,
                                            int col_base) {
  int lane = threadIdx.x & 31;
  int col  = col_base + (lane & 15);
  int kh   = (lane >> 4) << 3;
  v16bf f;
#pragma unroll
  for (int i = 0; i < 8; ++i) f[i] = base[(kh + i) * stride + col];
#pragma unroll
  for (int i = 0; i < 8; ++i) f[8 + i] = base[(16 + kh + i) * stride + col];
  return f;
}

// Shuffle reductions (wave32)
static __device__ inline float red_sum32(float v) {
#pragma unroll
  for (int m = 1; m < 32; m <<= 1) v += __shfl_xor(v, m, 32);
  return v;
}
static __device__ inline float red_sum16(float v) {
#pragma unroll
  for (int m = 1; m < 16; m <<= 1) v += __shfl_xor(v, m, 32);
  return v;
}
static __device__ inline float red_max16(float v) {
#pragma unroll
  for (int m = 1; m < 16; m <<= 1) v = fmaxf(v, __shfl_xor(v, m, 32));
  return v;
}

static __device__ inline float gelu_tanh(float x) {
  float x3 = x * x * x;
  return 0.5f * x * (1.0f + tanhf(0.79788456f * (x + 0.044715f * x3)));
}

// ---------------------------------------------------------------------------
// Problem constants
// ---------------------------------------------------------------------------
#define BB   16
#define SS   517      // 5 meta + 512 patches
#define RR   (BB*SS)  // 8272 rows
#define META 5
#define NP   512
#define WPP  128
#define DE   384
#define DD   512

// ---------------------------------------------------------------------------
// Generic WMMA GEMM:  C[M x N] = A[M x K] * W[N x K]^T  (+ epilogue)
// EPI bits: 1 = +bias[col], 2 = gelu, 4 = C = resid + g[col]*val
// block: 128 threads (4 waves); tile 64(M) x 64(N), K step 32.
// ---------------------------------------------------------------------------
template <int EPI>
__global__ __launch_bounds__(128) void gemm_kernel(
    const float* __restrict__ A, int lda, const float* __restrict__ W, int ldw,
    const float* __restrict__ bias, const float* __restrict__ resid,
    const float* __restrict__ gvec, float* __restrict__ C, int ldc, int M,
    int Nn, int K) {
  __shared__ alignas(16) __bf16 a_lds[64 * 40];
  __shared__ alignas(16) __bf16 w_lds[64 * 40];
  int tid = threadIdx.x, wv = tid >> 5, lane = tid & 31;
  int m0 = blockIdx.x * 64, n0 = blockIdx.y * 64;

  v8f acc[4];
#pragma unroll
  for (int c = 0; c < 4; ++c) acc[c] = zero8();

  for (int k0 = 0; k0 < K; k0 += 32) {
    // stage A (64x32) and W (64x32) tiles into LDS as bf16
#pragma unroll
    for (int ee = 0; ee < 16; ++ee) {
      int e = tid * 16 + ee;
      int r = e >> 5, kc = e & 31;
      int gr = m0 + r;
      if (gr >= M) gr = M - 1;
      a_lds[r * 40 + kc] = (__bf16)A[(size_t)gr * lda + k0 + kc];
      w_lds[r * 40 + kc] = (__bf16)W[(size_t)(n0 + r) * ldw + k0 + kc];
    }
    if (k0 + 32 < K) {
      int pr = m0 + (tid >> 1);
      if (pr >= M) pr = M - 1;
      __builtin_prefetch(A + (size_t)pr * lda + k0 + 32, 0, 1);
      __builtin_prefetch(W + (size_t)(n0 + (tid >> 1)) * ldw + k0 + 32, 0, 1);
    }
    __syncthreads();
    v16bf af = load_frag_rm(a_lds + wv * 16 * 40, 40);
#pragma unroll
    for (int c = 0; c < 4; ++c) {
      v16bf wf = load_frag_rm(w_lds + c * 16 * 40, 40);
      acc[c]   = wmma_bf16(af, wf, acc[c]);
    }
    __syncthreads();
  }

  int hi = lane >> 4, nlo = lane & 15;
#pragma unroll
  for (int c = 0; c < 4; ++c) {
#pragma unroll
    for (int j = 0; j < 8; ++j) {
      int row = m0 + wv * 16 + j + 8 * hi;
      int col = n0 + c * 16 + nlo;
      if (row < M) {
        float v = acc[c][j];
        if (EPI & 1) v += bias[col];
        if (EPI & 2) v = gelu_tanh(v);
        if (EPI & 4) v = resid[(size_t)row * ldc + col] + gvec[col] * v;
        C[(size_t)row * ldc + col] = v;
      }
    }
  }
}

// ---------------------------------------------------------------------------
// Patch embed as WMMA GEMM with gathered A: M = B*NP rows (one per patch),
// K = 256 pixels, N = 384. Epilogue: +patch_b +chan_table[ch], mask->0,
// scatter into h_enc[(b*S + meta + n), :].
// ---------------------------------------------------------------------------
__global__ __launch_bounds__(128) void patch_embed_kernel(
    const float* __restrict__ x, const float* __restrict__ pw,
    const float* __restrict__ pb, const float* __restrict__ ctab,
    const int* __restrict__ chans, const unsigned char* __restrict__ mask,
    float* __restrict__ h) {
  __shared__ alignas(16) __bf16 a_lds[64 * 40];
  __shared__ alignas(16) __bf16 w_lds[64 * 40];
  int tid = threadIdx.x, wv = tid >> 5, lane = tid & 31;
  int m0 = blockIdx.x * 64, n0 = blockIdx.y * 64;

  v8f acc[4];
#pragma unroll
  for (int c = 0; c < 4; ++c) acc[c] = zero8();

  for (int k0 = 0; k0 < 256; k0 += 32) {
#pragma unroll
    for (int ee = 0; ee < 16; ++ee) {
      int e = tid * 16 + ee;
      int r = e >> 5, kc = e & 31;
      int gr = m0 + r;           // patch row (b*512+n), M=8192 divisible by 64
      int kg = k0 + kc;
      int b = gr >> 9, n = gr & 511;
      int py = n >> 7, px = n & 127;
      int i = kg >> 4, j = kg & 15;
      a_lds[r * 40 + kc] =
          (__bf16)x[((size_t)b * 64 + py * 16 + i) * 2048 + px * 16 + j];
      w_lds[r * 40 + kc] = (__bf16)pw[(size_t)(n0 + r) * 256 + kg];
    }
    __syncthreads();
    v16bf af = load_frag_rm(a_lds + wv * 16 * 40, 40);
#pragma unroll
    for (int c = 0; c < 4; ++c) {
      v16bf wf = load_frag_rm(w_lds + c * 16 * 40, 40);
      acc[c]   = wmma_bf16(af, wf, acc[c]);
    }
    __syncthreads();
  }

  int hi = lane >> 4, nlo = lane & 15;
#pragma unroll
  for (int c = 0; c < 4; ++c) {
#pragma unroll
    for (int j = 0; j < 8; ++j) {
      int gr  = m0 + wv * 16 + j + 8 * hi;
      int col = n0 + c * 16 + nlo;
      int b = gr >> 9, n = gr & 511;
      float v = acc[c][j] + pb[col] + ctab[(size_t)chans[gr] * DE + col];
      if (mask[gr]) v = 0.f;
      h[((size_t)b * SS + META + n) * DE + col] = v;
    }
  }
}

// ---------------------------------------------------------------------------
// cls token + mean-token embed (tiny GEMM, VALU is fine: ~0.2 GFLOP total)
// ---------------------------------------------------------------------------
__global__ void meancls_kernel(const float* __restrict__ means,
                               const float* __restrict__ mw,
                               const float* __restrict__ mb,
                               const float* __restrict__ cls,
                               float* __restrict__ h) {
  int srow = blockIdx.x;  // 0..META-1
  int b    = blockIdx.y;
  float* dst = h + ((size_t)b * SS + srow) * DE;
  if (srow == 0) {
    for (int d = threadIdx.x; d < DE; d += blockDim.x) dst[d] = cls[d];
  } else {
    int mi = srow - 1;
    const float* mr = means + ((size_t)b * 4 + mi) * 4000;
    for (int d = threadIdx.x; d < DE; d += blockDim.x) {
      const float* wr = mw + (size_t)d * 4000;
      float s = mb[d];
      for (int k = 0; k < 4000; ++k) s += mr[k] * wr[k];
      dst[d] = s;
    }
  }
}

// ---------------------------------------------------------------------------
// LayerNorm: one wave per row (D = 384 or 512, both multiples of 32)
// ---------------------------------------------------------------------------
__global__ __launch_bounds__(128) void ln_kernel(const float* __restrict__ x,
                                                 float* __restrict__ y,
                                                 const float* __restrict__ w,
                                                 const float* __restrict__ b,
                                                 int R, int D) {
  int wv = threadIdx.x >> 5, lane = threadIdx.x & 31;
  int r = blockIdx.x * 4 + wv;
  if (r >= R) return;
  const float* xr = x + (size_t)r * D;
  float xv[16];
  int n = D / 32;
  float s = 0.f;
  for (int i = 0; i < n; ++i) {
    xv[i] = xr[lane + 32 * i];
    s += xv[i];
  }
  float mu = red_sum32(s) / (float)D;
  float v = 0.f;
  for (int i = 0; i < n; ++i) {
    float d = xv[i] - mu;
    v += d * d;
  }
  v = red_sum32(v) / (float)D;
  float rs = rsqrtf(v + 1e-6f);
  float* yr = y + (size_t)r * D;
  for (int i = 0; i < n; ++i) {
    int d = lane + 32 * i;
    yr[d] = (xv[i] - mu) * rs * w[d] + b[d];
  }
}

// ---------------------------------------------------------------------------
// Axial RoPE applied in-place to q and k inside the packed qkv buffer.
// One thread per rotation pair. Meta rows are identity (cos=1,sin=0).
// ---------------------------------------------------------------------------
__global__ void rope_kernel(float* __restrict__ qkv, int D, int dh,
                            float theta) {
  int half = D / 2;
  int gid = blockIdx.x * blockDim.x + threadIdx.x;
  if (gid >= RR * half) return;
  int row = gid / half, pr = gid % half;
  int s = row % SS;
  if (s < META) return;
  int h = pr / (dh / 2), i = pr % (dh / 2);
  int p = s - META;
  int nf = dh / 4;
  float pos = (i < nf) ? (float)(p % WPP) : (float)(p / WPP);
  int fi = (i < nf) ? i : i - nf;
  float freq = __powf(theta, -4.0f * (float)fi / (float)dh);
  float ang = pos * freq;
  float cs = __cosf(ang), sn = __sinf(ang);
  float* q = qkv + (size_t)row * 3 * D + h * dh + 2 * i;
  float* k = q + D;
  float q0 = q[0], q1 = q[1];
  q[0] = q0 * cs - q1 * sn;
  q[1] = q0 * sn + q1 * cs;
  float k0 = k[0], k1 = k[1];
  k[0] = k0 * cs - k1 * sn;
  k[1] = k0 * sn + k1 * cs;
}

// ---------------------------------------------------------------------------
// Flash attention, one wave per (b, head, 16-row q-tile).
// QK^T and PV both via WMMA; online softmax in f32 with 16-lane-group
// reductions that exactly match the C-matrix row striping (rows j / j+8).
// ---------------------------------------------------------------------------
template <int DH>
__global__ __launch_bounds__(32) void attn_kernel(const float* __restrict__ qkv,
                                                  float* __restrict__ out,
                                                  int D) {
  constexpr int NK = DH / 32;  // K-chunks for QK^T (dh=64 -> 2, dh=32 -> 1)
  constexpr int NC = DH / 16;  // 16-col chunks of O
  __shared__ alignas(16) __bf16 q_lds[16 * DH];
  __shared__ alignas(16) __bf16 k_lds[32 * DH];
  __shared__ alignas(16) __bf16 v_lds[32 * DH];
  __shared__ alignas(16) __bf16 p_lds[16 * 32];

  int lane = threadIdx.x;
  int qt = blockIdx.x, h = blockIdx.y, b = blockIdx.z;
  int D3 = 3 * D;
  const float* base = qkv + (size_t)b * SS * D3 + h * DH;

  for (int i = lane; i < 16 * DH; i += 32) {
    int r = i / DH, c = i % DH;
    int s = qt * 16 + r;
    if (s >= SS) s = SS - 1;
    q_lds[r * DH + c] = (__bf16)base[(size_t)s * D3 + c];
  }
  __syncthreads();
  v16bf aq[NK];
#pragma unroll
  for (int kk = 0; kk < NK; ++kk) aq[kk] = load_frag_rm(q_lds + kk * 32, DH);

  float mrow[8], lrow[8], alpha[8];
  v8f o[NC];
#pragma unroll
  for (int j = 0; j < 8; ++j) {
    mrow[j] = -3.0e38f;
    lrow[j] = 0.f;
  }
#pragma unroll
  for (int c = 0; c < NC; ++c) o[c] = zero8();
  float scale = rsqrtf((float)DH);
  int hi = lane >> 4, nlo = lane & 15;

  for (int t0 = 0; t0 < SS; t0 += 32) {
    {  // load K,V tiles: one row per lane
      int s = t0 + lane;
      if (s >= SS) s = SS - 1;
      const float* kp = base + (size_t)s * D3 + D;
      const float* vp = kp + D;
#pragma unroll
      for (int c = 0; c < DH; ++c) {
        k_lds[lane * DH + c] = (__bf16)kp[c];
        v_lds[lane * DH + c] = (__bf16)vp[c];
      }
    }
    __syncthreads();

    v8f sc[2];
    sc[0] = zero8();
    sc[1] = zero8();
#pragma unroll
    for (int kk = 0; kk < NK; ++kk) {
      v16bf bk0 = load_frag_rm(k_lds + kk * 32, DH);
      v16bf bk1 = load_frag_rm(k_lds + 16 * DH + kk * 32, DH);
      sc[0] = wmma_bf16(aq[kk], bk0, sc[0]);
      sc[1] = wmma_bf16(aq[kk], bk1, sc[1]);
    }

#pragma unroll
    for (int j = 0; j < 8; ++j) {
#pragma unroll
      for (int c = 0; c < 2; ++c) {
        int tg = t0 + c * 16 + nlo;
        float v = sc[c][j] * scale;
        if (tg >= SS) v = -3.0e38f;
        sc[c][j] = v;
      }
      float mx = red_max16(fmaxf(sc[0][j], sc[1][j]));
      float nm = fmaxf(mrow[j], mx);
      float al = __expf(mrow[j] - nm);
      float p0 = __expf(sc[0][j] - nm);
      float p1 = __expf(sc[1][j] - nm);
      sc[0][j] = p0;
      sc[1][j] = p1;
      lrow[j] = lrow[j] * al + red_sum16(p0 + p1);
      mrow[j] = nm;
      alpha[j] = al;
    }

    // re-layout P (C-layout -> A-layout) through LDS
#pragma unroll
    for (int j = 0; j < 8; ++j) {
      p_lds[(j + 8 * hi) * 32 + nlo]      = (__bf16)sc[0][j];
      p_lds[(j + 8 * hi) * 32 + 16 + nlo] = (__bf16)sc[1][j];
    }
    __syncthreads();
    v16bf ap = load_frag_rm(p_lds, 32);
#pragma unroll
    for (int c = 0; c < NC; ++c) {
#pragma unroll
      for (int j = 0; j < 8; ++j) o[c][j] *= alpha[j];
      v16bf bv = load_frag_cm(v_lds, DH, c * 16);
      o[c] = wmma_bf16(ap, bv, o[c]);
    }
    __syncthreads();
  }

#pragma unroll
  for (int c = 0; c < NC; ++c) {
#pragma unroll
    for (int j = 0; j < 8; ++j) {
      int s = qt * 16 + j + 8 * hi;
      if (s < SS) {
        out[((size_t)b * SS + s) * D + h * DH + c * 16 + nlo] =
            o[c][j] / lrow[j];
      }
    }
  }
}

// ---------------------------------------------------------------------------
// Loss: per-patch MSE vs the original pixels, masked mean (two-stage,
// deterministic). One wave per patch; 1024 blocks of 8 waves = 8192 patches.
// ---------------------------------------------------------------------------
__global__ __launch_bounds__(256) void loss_partial_kernel(
    const float* __restrict__ pred, const float* __restrict__ x,
    const unsigned char* __restrict__ mask, float* __restrict__ red) {
  __shared__ float s_s[8], s_c[8];
  int wv = threadIdx.x >> 5, lane = threadIdx.x & 31;
  int p = blockIdx.x * 8 + wv;  // 0..8191
  int b = p >> 9, n = p & 511;
  int py = n >> 7, px = n & 127;
  const float* pr = pred + ((size_t)b * SS + META + n) * 256;
  float ss = 0.f;
#pragma unroll
  for (int k = 0; k < 8; ++k) {
    int c = lane + 32 * k;
    int i = c >> 4, j = c & 15;
    float t = x[((size_t)b * 64 + py * 16 + i) * 2048 + px * 16 + j];
    float d = pr[c] - t;
    ss += d * d;
  }
  ss = red_sum32(ss) * (1.0f / 256.0f);
  float mf = mask[p] ? 1.f : 0.f;
  if (lane == 0) {
    s_s[wv] = ss * mf;
    s_c[wv] = mf;
  }
  __syncthreads();
  if (threadIdx.x == 0) {
    float a = 0.f, cn = 0.f;
    for (int i = 0; i < 8; ++i) {
      a += s_s[i];
      cn += s_c[i];
    }
    red[2 * blockIdx.x] = a;
    red[2 * blockIdx.x + 1] = cn;
  }
}

__global__ __launch_bounds__(256) void loss_final_kernel(
    const float* __restrict__ red, float* __restrict__ out, int npart) {
  __shared__ float ssa[8], ssc[8];
  float a = 0.f, c = 0.f;
  for (int i = threadIdx.x; i < npart; i += blockDim.x) {
    a += red[2 * i];
    c += red[2 * i + 1];
  }
  a = red_sum32(a);
  c = red_sum32(c);
  int wv = threadIdx.x >> 5, lane = threadIdx.x & 31;
  if (lane == 0) {
    ssa[wv] = a;
    ssc[wv] = c;
  }
  __syncthreads();
  if (threadIdx.x == 0) {
    float ta = 0.f, tc = 0.f;
    for (int i = 0; i < 8; ++i) {
      ta += ssa[i];
      tc += ssc[i];
    }
    out[0] = ta / tc;
  }
}

__global__ void pred_copy_kernel(const float* __restrict__ pred,
                                 float* __restrict__ out, int total) {
  int idx = blockIdx.x * blockDim.x + threadIdx.x;
  if (idx >= total) return;
  int p = idx >> 8, c = idx & 255;
  int b = p >> 9, n = p & 511;
  out[1 + idx] = pred[((size_t)b * SS + META + n) * 256 + c];
}

// ---------------------------------------------------------------------------
// Host-side orchestration
// ---------------------------------------------------------------------------
static void launch_gemm(int epi, const float* A, int lda, const float* W,
                        int ldw, const float* bias, const float* resid,
                        const float* g, float* C, int ldc, int M, int Nn, int K,
                        hipStream_t s) {
  dim3 grid((M + 63) / 64, Nn / 64);
  if (epi == 1)
    gemm_kernel<1><<<grid, 128, 0, s>>>(A, lda, W, ldw, bias, resid, g, C, ldc,
                                        M, Nn, K);
  else if (epi == 3)
    gemm_kernel<3><<<grid, 128, 0, s>>>(A, lda, W, ldw, bias, resid, g, C, ldc,
                                        M, Nn, K);
  else
    gemm_kernel<5><<<grid, 128, 0, s>>>(A, lda, W, ldw, bias, resid, g, C, ldc,
                                        M, Nn, K);
}

// input indices (setup_inputs dict order; nested param dicts in literal order)
enum {
  IN_X = 0, IN_MEANS, IN_CHANNELS, IN_MASK, IN_PATCH_W, IN_PATCH_B,
  IN_CHAN_TABLE, IN_MEAN_W, IN_MEAN_B, IN_CLS,
  ENC0 = 10, DEC0 = 24,
  IN_ENC_NORM_W = 38, IN_ENC_NORM_B, IN_DEC_EMBED_W, IN_DEC_EMBED_B,
  IN_DEC_NORM_W, IN_DEC_NORM_B, IN_PRED_W, IN_PRED_B
};
enum {
  P_QKV_W = 0, P_QKV_B, P_PROJ_W, P_PROJ_B, P_LN1_W, P_LN1_B, P_LN2_W, P_LN2_B,
  P_G1, P_G2, P_MLP_W1, P_MLP_B1, P_MLP_W2, P_MLP_B2
};

extern "C" void kernel_launch(void* const* d_in, const int* in_sizes, int n_in,
                              void* d_out, int out_size, void* d_ws,
                              size_t ws_size, hipStream_t stream) {
  (void)in_sizes; (void)n_in; (void)out_size; (void)ws_size;

  const float* xin      = (const float*)d_in[IN_X];
  const float* means    = (const float*)d_in[IN_MEANS];
  const int*   channels = (const int*)d_in[IN_CHANNELS];
  const unsigned char* mask = (const unsigned char*)d_in[IN_MASK];
  const float* patch_w  = (const float*)d_in[IN_PATCH_W];
  const float* patch_b  = (const float*)d_in[IN_PATCH_B];
  const float* ctab     = (const float*)d_in[IN_CHAN_TABLE];
  const float* mean_w   = (const float*)d_in[IN_MEAN_W];
  const float* mean_b   = (const float*)d_in[IN_MEAN_B];
  const float* cls      = (const float*)d_in[IN_CLS];
  const float* enc[14], *dec[14];
  for (int i = 0; i < 14; ++i) {
    enc[i] = (const float*)d_in[ENC0 + i];
    dec[i] = (const float*)d_in[DEC0 + i];
  }
  const float* enc_nw = (const float*)d_in[IN_ENC_NORM_W];
  const float* enc_nb = (const float*)d_in[IN_ENC_NORM_B];
  const float* dembw  = (const float*)d_in[IN_DEC_EMBED_W];
  const float* dembb  = (const float*)d_in[IN_DEC_EMBED_B];
  const float* dec_nw = (const float*)d_in[IN_DEC_NORM_W];
  const float* dec_nb = (const float*)d_in[IN_DEC_NORM_B];
  const float* pred_w = (const float*)d_in[IN_PRED_W];
  const float* pred_b = (const float*)d_in[IN_PRED_B];
  float* out = (float*)d_out;

  // workspace layout (fp32)
  float* ws = (float*)d_ws;
  size_t o = 0;
  float* h_enc   = ws + o; o += (size_t)RR * DE;
  float* h_dec   = ws + o; o += (size_t)RR * DD;
  float* lnbuf   = ws + o; o += (size_t)RR * DD;
  float* qkvbuf  = ws + o; o += (size_t)RR * 3 * DD;
  float* attnout = ws + o; o += (size_t)RR * DD;
  float* midbuf  = ws + o; o += (size_t)RR * 2048;  // also reused for pred
  float* redbuf  = ws + o; o += 2048;

  const int lnBlocks = (RR + 3) / 4;

  // ---- embed: cls + mean tokens + patch tokens -> h_enc (B,S,384)
  meancls_kernel<<<dim3(META, BB), 128, 0, stream>>>(means, mean_w, mean_b,
                                                     cls, h_enc);
  patch_embed_kernel<<<dim3((BB * NP) / 64, DE / 64), 128, 0, stream>>>(
      xin, patch_w, patch_b, ctab, channels, mask, h_enc);

  // ---- encoder: 12 blocks, D=384, 6 heads dh=64
  for (int l = 0; l < 12; ++l) {
    ln_kernel<<<lnBlocks, 128, 0, stream>>>(h_enc, lnbuf, enc[P_LN1_W] + l * DE,
                                            enc[P_LN1_B] + l * DE, RR, DE);
    launch_gemm(1, lnbuf, DE, enc[P_QKV_W] + (size_t)l * 3 * DE * DE, DE,
                enc[P_QKV_B] + (size_t)l * 3 * DE, nullptr, nullptr, qkvbuf,
                3 * DE, RR, 3 * DE, DE, stream);
    rope_kernel<<<(RR * (DE / 2) + 255) / 256, 256, 0, stream>>>(qkvbuf, DE, 64,
                                                                 100.0f);
    attn_kernel<64><<<dim3((SS + 15) / 16, 6, BB), 32, 0, stream>>>(qkvbuf,
                                                                    attnout, DE);
    launch_gemm(5, attnout, DE, enc[P_PROJ_W] + (size_t)l * DE * DE, DE,
                enc[P_PROJ_B] + (size_t)l * DE, h_enc, enc[P_G1] + l * DE,
                h_enc, DE, RR, DE, DE, stream);
    ln_kernel<<<lnBlocks, 128, 0, stream>>>(h_enc, lnbuf, enc[P_LN2_W] + l * DE,
                                            enc[P_LN2_B] + l * DE, RR, DE);
    launch_gemm(3, lnbuf, DE, enc[P_MLP_W1] + (size_t)l * 4 * DE * DE, DE,
                enc[P_MLP_B1] + (size_t)l * 4 * DE, nullptr, nullptr, midbuf,
                4 * DE, RR, 4 * DE, DE, stream);
    launch_gemm(5, midbuf, 4 * DE, enc[P_MLP_W2] + (size_t)l * DE * 4 * DE,
                4 * DE, enc[P_MLP_B2] + (size_t)l * DE, h_enc,
                enc[P_G2] + l * DE, h_enc, DE, RR, DE, 4 * DE, stream);
  }

  // ---- encoder norm + decoder embed (384 -> 512)
  ln_kernel<<<lnBlocks, 128, 0, stream>>>(h_enc, lnbuf, enc_nw, enc_nb, RR, DE);
  launch_gemm(1, lnbuf, DE, dembw, DE, dembb, nullptr, nullptr, h_dec, DD, RR,
              DD, DE, stream);

  // ---- decoder: 8 blocks, D=512, 16 heads dh=32
  for (int l = 0; l < 8; ++l) {
    ln_kernel<<<lnBlocks, 128, 0, stream>>>(h_dec, lnbuf, dec[P_LN1_W] + l * DD,
                                            dec[P_LN1_B] + l * DD, RR, DD);
    launch_gemm(1, lnbuf, DD, dec[P_QKV_W] + (size_t)l * 3 * DD * DD, DD,
                dec[P_QKV_B] + (size_t)l * 3 * DD, nullptr, nullptr, qkvbuf,
                3 * DD, RR, 3 * DD, DD, stream);
    rope_kernel<<<(RR * (DD / 2) + 255) / 256, 256, 0, stream>>>(qkvbuf, DD, 32,
                                                                 100.0f);
    attn_kernel<32><<<dim3((SS + 15) / 16, 16, BB), 32, 0, stream>>>(
        qkvbuf, attnout, DD);
    launch_gemm(5, attnout, DD, dec[P_PROJ_W] + (size_t)l * DD * DD, DD,
                dec[P_PROJ_B] + (size_t)l * DD, h_dec, dec[P_G1] + l * DD,
                h_dec, DD, RR, DD, DD, stream);
    ln_kernel<<<lnBlocks, 128, 0, stream>>>(h_dec, lnbuf, dec[P_LN2_W] + l * DD,
                                            dec[P_LN2_B] + l * DD, RR, DD);
    launch_gemm(3, lnbuf, DD, dec[P_MLP_W1] + (size_t)l * 4 * DD * DD, DD,
                dec[P_MLP_B1] + (size_t)l * 4 * DD, nullptr, nullptr, midbuf,
                4 * DD, RR, 4 * DD, DD, stream);
    launch_gemm(5, midbuf, 4 * DD, dec[P_MLP_W2] + (size_t)l * DD * 4 * DD,
                4 * DD, dec[P_MLP_B2] + (size_t)l * DD, h_dec,
                dec[P_G2] + l * DD, h_dec, DD, RR, DD, 4 * DD, stream);
  }

  // ---- decoder norm + prediction head (512 -> 256)
  ln_kernel<<<lnBlocks, 128, 0, stream>>>(h_dec, lnbuf, dec_nw, dec_nb, RR, DD);
  float* predbuf = midbuf;  // reuse (R x 256)
  launch_gemm(1, lnbuf, DD, pred_w, DD, pred_b, nullptr, nullptr, predbuf, 256,
              RR, 256, DD, stream);

  // ---- masked MSE loss + pred output
  loss_partial_kernel<<<1024, 256, 0, stream>>>(predbuf, xin, mask, redbuf);
  loss_final_kernel<<<1, 256, 0, stream>>>(redbuf, out, 1024);
  int total = BB * NP * 256;
  pred_copy_kernel<<<(total + 255) / 256, 256, 0, stream>>>(predbuf, out,
                                                            total);
}
